// FlashAttentionLinks_79783312491269
// MI455X (gfx1250) — matrix operand order repaired and checked by reading, not compiled, not run on hardware
//
#include <hip/hip_runtime.h>

// ---------------------------------------------------------------------------
// Problem constants (match reference)
// ---------------------------------------------------------------------------
#define BB   2
#define LL   2048
#define EE   768
#define DD   512
#define EPSF 1e-6f
#define LN_EPSF 1e-5f
#define NEG_BIGF -1e9f

typedef __attribute__((ext_vector_type(16))) __bf16 v16bf;
typedef __attribute__((ext_vector_type(8)))  __bf16 v8bf;
typedef __attribute__((ext_vector_type(8)))  float  v8f;

// ---------------------------------------------------------------------------
// WMMA wrapper: D = A(16x32 bf16) * B(32x16 bf16) + C(f32)
// ---------------------------------------------------------------------------
__device__ __forceinline__ v8f wmma_bf16(v16bf a, v16bf b, v8f c) {
  return __builtin_amdgcn_wmma_f32_16x16x32_bf16(
      /*neg_a=*/false, a, /*neg_b=*/false, b,
      /*c_mod=*/(short)0, c, /*reuse_a=*/false, /*reuse_b=*/false);
}

__device__ __forceinline__ v16bf join16(v8bf lo, v8bf hi) {
  return __builtin_shufflevector(lo, hi, 0, 1, 2, 3, 4, 5, 6, 7, 8, 9, 10, 11,
                                 12, 13, 14, 15);
}

// A fragment (16x32 bf16, wave32), K contiguous (sAk==1).
// lane l: row m = m0 + (l&15); lanes 0-15 take K=[0..7]+[16..23],
// lanes 16-31 take K=[8..15]+[24..31].  Two aligned 16B loads.
__device__ __forceinline__ v16bf load_frag_A(const __bf16* __restrict__ A,
                                             long long sAm, int m0, int k0,
                                             int lane) {
  const __bf16* base =
      A + (long long)(m0 + (lane & 15)) * sAm + k0 + (lane >> 4) * 8;
  v8bf lo = *(const v8bf*)(base);
  v8bf hi = *(const v8bf*)(base + 16);
  return join16(lo, hi);
}

// B fragment (32x16 bf16, wave32), K contiguous (sBk==1).
// lane l: col n = n0 + (l&15); lanes 0-15 hold K=[0..7]+[16..23], lanes 16-31 K=[8..15]+[24..31].
__device__ __forceinline__ v16bf load_frag_B(const __bf16* __restrict__ Bm,
                                             long long sBn, int k0, int n0,
                                             int lane) {
  const __bf16* base =
      Bm + (long long)(n0 + (lane & 15)) * sBn + k0 + (lane >> 4) * 8;
  v8bf lo = *(const v8bf*)(base);
  v8bf hi = *(const v8bf*)(base + 16);
  return join16(lo, hi);
}

// ---------------------------------------------------------------------------
// Batched bf16 GEMM, unit K stride both operands:
//   C[m,n] = sum_k A[m*sAm + k] * B[n*sBn + k]   (f32 accumulate)
// Block: 128 threads = 4 waves. Wave tile 32(M)x64(N); block tile 128x64.
// Grid: (N/64, M/128, nbatch). K multiple of 32.
// ---------------------------------------------------------------------------
__global__ void gemm_bf16_kernel(const __bf16* __restrict__ A,
                                 const __bf16* __restrict__ Bm,
                                 float* __restrict__ C, int K, long long sAm,
                                 long long sBn, long long ldc, long long bA,
                                 long long bB, long long bC) {
  const int lane = threadIdx.x & 31;
  const int wave = threadIdx.x >> 5;
  const long long zb = blockIdx.z;
  const __bf16* Ab = A + zb * bA;
  const __bf16* Bb = Bm + zb * bB;
  float* Cb = C + zb * bC;

  const int m0 = blockIdx.y * 128 + wave * 32;  // wave owns rows m0..m0+31
  const int n0 = blockIdx.x * 64;

  v8f acc[2][4];
#pragma unroll
  for (int i = 0; i < 2; ++i)
#pragma unroll
    for (int j = 0; j < 4; ++j) acc[i][j] = (v8f){};

  for (int k0 = 0; k0 < K; k0 += 32) {
    v16bf a0 = load_frag_A(Ab, sAm, m0, k0, lane);
    v16bf a1 = load_frag_A(Ab, sAm, m0 + 16, k0, lane);
    v16bf b0 = load_frag_B(Bb, sBn, k0, n0 + 0, lane);
    v16bf b1 = load_frag_B(Bb, sBn, k0, n0 + 16, lane);
    v16bf b2 = load_frag_B(Bb, sBn, k0, n0 + 32, lane);
    v16bf b3 = load_frag_B(Bb, sBn, k0, n0 + 48, lane);
    acc[0][0] = wmma_bf16(a0, b0, acc[0][0]);
    acc[0][1] = wmma_bf16(a0, b1, acc[0][1]);
    acc[0][2] = wmma_bf16(a0, b2, acc[0][2]);
    acc[0][3] = wmma_bf16(a0, b3, acc[0][3]);
    acc[1][0] = wmma_bf16(a1, b0, acc[1][0]);
    acc[1][1] = wmma_bf16(a1, b1, acc[1][1]);
    acc[1][2] = wmma_bf16(a1, b2, acc[1][2]);
    acc[1][3] = wmma_bf16(a1, b3, acc[1][3]);
  }

  // D layout: element r of lane l -> row = (l>>4)*8 + r, col = l&15
  const int rbase = m0 + (lane >> 4) * 8;
  const int cb = n0 + (lane & 15);
#pragma unroll
  for (int i = 0; i < 2; ++i) {
#pragma unroll
    for (int r = 0; r < 8; ++r) {
      long long rr = rbase + i * 16 + r;
      Cb[rr * ldc + cb + 0]  = acc[i][0][r];
      Cb[rr * ldc + cb + 16] = acc[i][1][r];
      Cb[rr * ldc + cb + 32] = acc[i][2][r];
      Cb[rr * ldc + cb + 48] = acc[i][3][r];
    }
  }
}

// ---------------------------------------------------------------------------
// f32 -> bf16 conversion (row-major copy)
// ---------------------------------------------------------------------------
__global__ void cvt_f32_bf16(__bf16* __restrict__ dst,
                             const float* __restrict__ src, long long n) {
  long long i = (long long)blockIdx.x * blockDim.x + threadIdx.x;
  long long stride = (long long)gridDim.x * blockDim.x;
  for (; i < n; i += stride) dst[i] = (__bf16)src[i];
}

// Tiled per-batch transpose: dst[c*R + r] = src[r*C + c]. grid(C/32, R/32, nb)
__global__ void transpose_bf16_kernel(__bf16* __restrict__ dst,
                                      const __bf16* __restrict__ src, int R,
                                      int C) {
  __shared__ __bf16 tile[32][33];
  const long long zb = blockIdx.z;
  const __bf16* s = src + zb * (long long)R * C;
  __bf16* d = dst + zb * (long long)R * C;
  int bc = blockIdx.x * 32, br = blockIdx.y * 32;
  int tx = threadIdx.x, ty = threadIdx.y;  // 32 x 8
  for (int i = ty; i < 32; i += 8)
    tile[i][tx] = s[(long long)(br + i) * C + bc + tx];
  __syncthreads();
  for (int i = ty; i < 32; i += 8)
    d[(long long)(bc + i) * R + br + tx] = tile[tx][i];
}

// Tiled per-batch transpose + convert: dst(bf16)[c*R+r] = src(f32)[r*C+c]
__global__ void transpose_cvt_f32_bf16_kernel(__bf16* __restrict__ dst,
                                              const float* __restrict__ src,
                                              int R, int C) {
  __shared__ float tile[32][33];
  const long long zb = blockIdx.z;
  const float* s = src + zb * (long long)R * C;
  __bf16* d = dst + zb * (long long)R * C;
  int bc = blockIdx.x * 32, br = blockIdx.y * 32;
  int tx = threadIdx.x, ty = threadIdx.y;
  for (int i = ty; i < 32; i += 8)
    tile[i][tx] = s[(long long)(br + i) * C + bc + tx];
  __syncthreads();
  for (int i = ty; i < 32; i += 8)
    d[(long long)(bc + i) * R + br + tx] = (__bf16)tile[tx][i];
}

// ---------------------------------------------------------------------------
// Fused LayerNorm (over D=512) + RoPE; writes bf16. One block per row.
// ---------------------------------------------------------------------------
__global__ void ln_rope_kernel(__bf16* __restrict__ out,
                               const float* __restrict__ raw,
                               const float* __restrict__ gamma,
                               const float* __restrict__ beta) {
  __shared__ float red[256];
  const int row = blockIdx.x;
  const int tid = threadIdx.x;
  const float* r = raw + (long long)row * DD;
  const int d1 = tid, d2 = tid + 256;

  float v1 = r[d1], v2 = r[d2];
  red[tid] = v1 + v2;
  __syncthreads();
  for (int s = 128; s > 0; s >>= 1) {
    if (tid < s) red[tid] += red[tid + s];
    __syncthreads();
  }
  float mu = red[0] * (1.0f / DD);
  __syncthreads();

  float c1 = v1 - mu, c2 = v2 - mu;
  red[tid] = c1 * c1 + c2 * c2;
  __syncthreads();
  for (int s = 128; s > 0; s >>= 1) {
    if (tid < s) red[tid] += red[tid + s];
    __syncthreads();
  }
  float var = red[0] * (1.0f / DD);
  float inv = rsqrtf(var + LN_EPSF);

  float n1 = c1 * inv * gamma[d1] + beta[d1];
  float n2 = c2 * inv * gamma[d2] + beta[d2];

  const int l = row & (LL - 1);
  float invf = expf(-(2.0f * (float)tid / (float)DD) * logf(10000.0f));
  float ang = (float)l * invf;
  float cs = cosf(ang), sn = sinf(ang);

  float o1 = n1 * cs - n2 * sn;
  float o2 = n2 * cs + n1 * sn;

  out[(long long)row * DD + d1] = (__bf16)o1;
  out[(long long)row * DD + d2] = (__bf16)o2;
}

// ---------------------------------------------------------------------------
// Exact 1.5-entmax over a 2048-wide row held in LDS.
// ---------------------------------------------------------------------------
struct EntmaxSmem {
  float z[LL];
  float zs[LL];
  float cum[LL];
  float cum2[LL];
  float small[256];
  float small2[256];
};

__device__ void entmax15_row(EntmaxSmem& sm, float* __restrict__ outRow,
                             int tid, bool clipUpper, bool affine) {
  const int T = 256;

  float lmax = -3.0e38f;
  for (int j = tid; j < LL; j += T) {
    float v = sm.z[j] * 0.5f;
    sm.z[j] = v;
    lmax = fmaxf(lmax, v);
  }
  sm.small[tid] = lmax;
  __syncthreads();
  for (int s = 128; s > 0; s >>= 1) {
    if (tid < s) sm.small[tid] = fmaxf(sm.small[tid], sm.small[tid + s]);
    __syncthreads();
  }
  float zmax = sm.small[0];
  __syncthreads();

  for (int j = tid; j < LL; j += T) {
    float v = sm.z[j] - zmax;
    sm.z[j] = v;
    sm.zs[j] = v;
  }
  __syncthreads();

  // Bitonic sort, descending
  for (unsigned k = 2; k <= LL; k <<= 1) {
    for (unsigned j = k >> 1; j > 0; j >>= 1) {
      for (unsigned idx = tid; idx < LL; idx += T) {
        unsigned ixj = idx ^ j;
        if (ixj > idx) {
          float a = sm.zs[idx], b = sm.zs[ixj];
          bool up = ((idx & k) == 0);
          bool doswap = up ? (a < b) : (a > b);
          if (doswap) { sm.zs[idx] = b; sm.zs[ixj] = a; }
        }
      }
      __syncthreads();
    }
  }

  // Chunked serial scans + block scan of chunk totals
  float lc[8], lc2[8];
  float run = 0.0f, run2 = 0.0f;
  const int base = tid * 8;
#pragma unroll
  for (int e = 0; e < 8; ++e) {
    float v = sm.zs[base + e];
    run += v;
    run2 += v * v;
    lc[e] = run;
    lc2[e] = run2;
  }
  sm.small[tid] = run;
  sm.small2[tid] = run2;
  __syncthreads();
  for (int off = 1; off < T; off <<= 1) {
    float a = sm.small[tid], a2 = sm.small2[tid];
    float b = 0.0f, b2 = 0.0f;
    if (tid >= off) { b = sm.small[tid - off]; b2 = sm.small2[tid - off]; }
    __syncthreads();
    sm.small[tid] = a + b;
    sm.small2[tid] = a2 + b2;
    __syncthreads();
  }
  float pre  = (tid == 0) ? 0.0f : sm.small[tid - 1];
  float pre2 = (tid == 0) ? 0.0f : sm.small2[tid - 1];
  __syncthreads();

  float lsupport = 0.0f;
#pragma unroll
  for (int e = 0; e < 8; ++e) {
    float cum = pre + lc[e];
    float cum2 = pre2 + lc2[e];
    sm.cum[base + e] = cum;
    sm.cum2[base + e] = cum2;
    float kk = (float)(base + e + 1);
    float meanv = cum / kk;
    float ssv = cum2 - cum * meanv;
    float delta = (1.0f - ssv) / kk;
    float tau = meanv - sqrtf(fmaxf(delta, 0.0f));
    lsupport += (tau <= sm.zs[base + e]) ? 1.0f : 0.0f;
  }
  __syncthreads();
  sm.small[tid] = lsupport;
  __syncthreads();
  for (int s = 128; s > 0; s >>= 1) {
    if (tid < s) sm.small[tid] += sm.small[tid + s];
    __syncthreads();
  }
  int sup = (int)(sm.small[0] + 0.5f);
  if (sup < 1) sup = 1;
  if (sup > LL) sup = LL;
  __syncthreads();
  if (tid == 0) {
    float cum = sm.cum[sup - 1], cum2 = sm.cum2[sup - 1];
    float kk = (float)sup;
    float meanv = cum / kk;
    float ssv = cum2 - cum * meanv;
    float delta = (1.0f - ssv) / kk;
    sm.small[0] = meanv - sqrtf(fmaxf(delta, 0.0f));
  }
  __syncthreads();
  float tau_star = sm.small[0];

  for (int j = tid; j < LL; j += T) {
    float p = fmaxf(sm.z[j] - tau_star, 0.0f);
    p = p * p;
    if (clipUpper) p = fminf(p, 1.0f - EPSF);
    if (affine) p = fmaxf((1.0f - 2.0f * EPSF) * p + EPSF, EPSF);
    outRow[j] = p;
  }
}

__global__ void entmax_rows_kernel(float* __restrict__ outP,
                                   const float* __restrict__ raw,
                                   const float* __restrict__ wptr,
                                   int affine) {
  __shared__ EntmaxSmem sm;
  const int row = blockIdx.x;
  const int tid = threadIdx.x;
  const float* r = raw + (long long)row * LL;

  float lsum = 0.0f;
  for (int j = tid; j < LL; j += 256) lsum += r[j];
  sm.small[tid] = lsum;
  __syncthreads();
  for (int s = 128; s > 0; s >>= 1) {
    if (tid < s) sm.small[tid] += sm.small[tid + s];
    __syncthreads();
  }
  float mean = sm.small[0] * (1.0f / LL);
  __syncthreads();

  float w = 2.0f / (1.0f + expf(-wptr[0]));
  for (int j = tid; j < LL; j += 256) sm.z[j] = w * (r[j] - mean);
  __syncthreads();

  entmax15_row(sm, outP + (long long)row * LL, tid, /*clipUpper=*/true,
               affine != 0);
}

__global__ void colsum_kernel(float* __restrict__ cs,
                              const float* __restrict__ p) {
  int g = blockIdx.x * blockDim.x + threadIdx.x;
  if (g >= BB * LL) return;
  int b = g / LL, j = g % LL;
  const float* base = p + (long long)b * LL * LL + j;
  float s = 0.0f;
  for (int i = 0; i < LL; ++i) s += base[(long long)i * LL];
  cs[g] = s;
}

__global__ void scale_affine_kernel(float* __restrict__ p,
                                    const float* __restrict__ cs) {
  long long g = (long long)blockIdx.x * blockDim.x + threadIdx.x;
  long long n = (long long)BB * LL * LL;
  if (g >= n) return;
  long long b = g / ((long long)LL * LL);
  long long j = g % LL;
  float r = rsqrtf(cs[b * LL + j] + EPSF);
  float v = p[g] * r;
  p[g] = fmaxf((1.0f - 2.0f * EPSF) * v + EPSF, EPSF);
}

__global__ void entmax_H_kernel(float* __restrict__ outH,
                                const float* __restrict__ pC,
                                const float* __restrict__ pF,
                                const float* __restrict__ pS) {
  __shared__ EntmaxSmem sm;
  const int row = blockIdx.x;
  const int tid = threadIdx.x;
  const int i = row & (LL - 1);
  const long long base = (long long)row * LL;
  for (int j = tid; j < LL; j += 256) {
    float v;
    if (j == i) {
      v = NEG_BIGF;
    } else {
      v = (logf(pC[base + j]) + logf(pF[base + j]) + logf(pS[base + j])) *
          (1.0f / 3.0f);
    }
    sm.z[j] = v;
  }
  __syncthreads();
  entmax15_row(sm, outH + base, tid, /*clipUpper=*/false, /*affine=*/false);
}

// ---------------------------------------------------------------------------
// Host-side launch
// ---------------------------------------------------------------------------
static inline void* ws_take(char*& cur, size_t bytes) {
  void* p = (void*)cur;
  cur += (bytes + 255) & ~(size_t)255;
  return p;
}

extern "C" void kernel_launch(void* const* d_in, const int* in_sizes, int n_in,
                              void* d_out, int out_size, void* d_ws,
                              size_t ws_size, hipStream_t stream) {
  (void)in_sizes; (void)n_in; (void)out_size; (void)ws_size;

  const float* x  = (const float*)d_in[0];  // [B,L,E]
  const float* Wq = (const float*)d_in[1];  // [D,E]
  const float* Wk = (const float*)d_in[2];  // [D,E]
  const float* qg = (const float*)d_in[3];
  const float* qb = (const float*)d_in[4];
  const float* kg = (const float*)d_in[5];
  const float* kb = (const float*)d_in[6];
  const float* Cw = (const float*)d_in[7];
  const float* Fw = (const float*)d_in[8];
  const float* Sw = (const float*)d_in[9];

  const long long M   = (long long)BB * LL;       // 4096 rows total
  const long long BLL = (long long)BB * LL * LL;

  float* outH = (float*)d_out;
  float* outC = outH + BLL;
  float* outF = outC + BLL;
  float* outS = outF + BLL;

  // Workspace carve-up
  char* cur = (char*)d_ws;
  __bf16* xb  = (__bf16*)ws_take(cur, (size_t)M * EE * 2);        // 6 MiB
  __bf16* wqb = (__bf16*)ws_take(cur, (size_t)DD * EE * 2);       // .75 MiB
  __bf16* wkb = (__bf16*)ws_take(cur, (size_t)DD * EE * 2);       // .75 MiB
  float*  praw= (float*) ws_take(cur, (size_t)M * DD * 4);        // 8 MiB
  __bf16* qbf = (__bf16*)ws_take(cur, (size_t)M * DD * 2);        // 4 MiB
  __bf16* kbf = (__bf16*)ws_take(cur, (size_t)M * DD * 2);        // 4 MiB
  __bf16* qT  = (__bf16*)ws_take(cur, (size_t)M * DD * 2);        // 4 MiB
  __bf16* kT  = (__bf16*)ws_take(cur, (size_t)M * DD * 2);        // 4 MiB
  float*  gf  = (float*) ws_take(cur, (size_t)BB * DD * DD * 4);  // 2 MiB
  __bf16* gT  = (__bf16*)ws_take(cur, (size_t)BB * DD * DD * 2);  // 1 MiB
  float*  tf  = (float*) ws_take(cur, (size_t)M * DD * 4);        // 8 MiB
  __bf16* tbf = (__bf16*)ws_take(cur, (size_t)M * DD * 2);        // 4 MiB
  float*  raw = (float*) ws_take(cur, (size_t)BLL * 4);           // 32 MiB
  float*  cs  = (float*) ws_take(cur, (size_t)BB * LL * 4);       // 16 KiB

  // ---- convert inputs to bf16 ----
  {
    long long n = M * EE;
    cvt_f32_bf16<<<(unsigned)((n + 255) / 256), 256, 0, stream>>>(xb, x, n);
    n = (long long)DD * EE;
    cvt_f32_bf16<<<(unsigned)((n + 255) / 256), 256, 0, stream>>>(wqb, Wq, n);
    cvt_f32_bf16<<<(unsigned)((n + 255) / 256), 256, 0, stream>>>(wkb, Wk, n);
  }

  const dim3 tb(32, 8, 1);

  // ---- projections + LN + RoPE + per-batch transpose ----
  // q = x @ Wq^T: A=xb (sAm=E), B=Wq row-major (sBn=E)
  gemm_bf16_kernel<<<dim3(DD / 64, (unsigned)(M / 128), 1), 128, 0, stream>>>(
      xb, wqb, praw, EE, EE, EE, DD, 0, 0, 0);
  ln_rope_kernel<<<(unsigned)M, 256, 0, stream>>>(qbf, praw, qg, qb);
  transpose_bf16_kernel<<<dim3(DD / 32, LL / 32, BB), tb, 0, stream>>>(
      qT, qbf, LL, DD);

  gemm_bf16_kernel<<<dim3(DD / 64, (unsigned)(M / 128), 1), 128, 0, stream>>>(
      xb, wkb, praw, EE, EE, EE, DD, 0, 0, 0);
  ln_rope_kernel<<<(unsigned)M, 256, 0, stream>>>(kbf, praw, kg, kb);
  transpose_bf16_kernel<<<dim3(DD / 32, LL / 32, BB), tb, 0, stream>>>(
      kT, kbf, LL, DD);

  // ---- three score matrices, sequential to reuse `raw` ----
  struct PhaseDesc {
    const __bf16* gAT;  // Gram A, [D,L] transposed copy
    const __bf16* gBT;  // Gram B, [D,L] transposed copy
    const __bf16* tA;   // left of T = tA @ G, row-major [L,D]
    const __bf16* rB;   // right: raw = T @ rB^T, row-major [L,D]
    const float*  w;
    float*        out;
    int           needs_colsum;
  } ph[3] = {
      {kT, kT, qbf, qbf, Cw, outC, 1},  // C: q Gkk q^T
      {kT, qT, qbf, kbf, Fw, outF, 0},  // F: q Gkq k^T
      {qT, qT, kbf, kbf, Sw, outS, 1},  // S: k Gqq k^T
  };

  const long long bLD  = (long long)LL * DD;
  const long long bDD  = (long long)DD * DD;
  const long long bLLs = (long long)LL * LL;

  for (int p = 0; p < 3; ++p) {
    // Gram: G[d,e] = sum_l gAT[d*L+l]*gBT[e*L+l]  (per batch)
    gemm_bf16_kernel<<<dim3(DD / 64, DD / 128, BB), 128, 0, stream>>>(
        ph[p].gAT, ph[p].gBT, gf, LL, /*sAm=*/LL, /*sBn=*/LL, /*ldc=*/DD,
        bLD, bLD, bDD);
    // G^T (bf16) so the T-GEMM's B operand is K-contiguous
    transpose_cvt_f32_bf16_kernel<<<dim3(DD / 32, DD / 32, BB), tb, 0,
                                    stream>>>(gT, gf, DD, DD);
    // T = tA @ G: B(k,n)=G[k*D+n]=gT[n*D+k]  (sBn=D)
    gemm_bf16_kernel<<<dim3(DD / 64, LL / 128, BB), 128, 0, stream>>>(
        ph[p].tA, gT, tf, DD, /*sAm=*/DD, /*sBn=*/DD, /*ldc=*/DD,
        bLD, bDD, bLD);
    {
      long long n = M * DD;
      cvt_f32_bf16<<<(unsigned)((n + 255) / 256), 256, 0, stream>>>(tbf, tf, n);
    }
    // raw = T @ rB^T: B(k=d,n=j)=rB[j*D+d]  (sBn=D)
    gemm_bf16_kernel<<<dim3(LL / 64, LL / 128, BB), 128, 0, stream>>>(
        tbf, ph[p].rB, raw, DD, /*sAm=*/DD, /*sBn=*/DD, /*ldc=*/LL,
        bLD, bLD, bLLs);

    entmax_rows_kernel<<<(unsigned)M, 256, 0, stream>>>(
        ph[p].out, raw, ph[p].w, /*affine=*/ph[p].needs_colsum ? 0 : 1);

    if (ph[p].needs_colsum) {
      colsum_kernel<<<(unsigned)((BB * LL + 255) / 256), 256, 0, stream>>>(
          cs, ph[p].out);
      long long n = BLL;
      scale_affine_kernel<<<(unsigned)((n + 255) / 256), 256, 0, stream>>>(
          ph[p].out, cs);
    }
  }

  // ---- final H ----
  entmax_H_kernel<<<(unsigned)M, 256, 0, stream>>>(outH, outC, outF, outS);
}
